// AsynchronousGRUCritic_60722247631601
// MI455X (gfx1250) — matrix operand order repaired and checked by reading, not compiled, not run on hardware
//
#include <hip/hip_runtime.h>

typedef __bf16 bf16_t;
typedef __attribute__((ext_vector_type(16))) __bf16 v16bf;
typedef __attribute__((ext_vector_type(8)))  __bf16 v8bf;
typedef __attribute__((ext_vector_type(8)))  float  v8f;
typedef __attribute__((ext_vector_type(4)))  float  v4f;

#define B_TOTAL 262144
#define D_IN    256
#define H_DIM   64

#define WAVES          8
#define THREADS        (WAVES * 32)
#define ROWS_PER_BLOCK (WAVES * 16)

// LDS weight fragment buffer (phased use):
//  Phase fc1: stream0 frags (16384 bf16) + stream1 frags (16384) = 64 KB
//  Phase gru: ih_s0(12288) ih_s1(12288) hh_s0(12288) hh_s1(12288) = 96 KB
#define WFRAG_ELEMS 49152
#define XH_STRIDE   72   // padded row stride (bf16 elems) to reduce LDS bank conflicts

__device__ __forceinline__ float sigmoid_f(float x) { return 1.0f / (1.0f + __expf(-x)); }
__device__ __forceinline__ float tanh_f(float x)    { return 1.0f - 2.0f / (__expf(2.0f * x) + 1.0f); }

// Convert f32 weight W[N][K] (row-major) into bf16 B-fragments for v_wmma_*_bf16.
// Fragment order: frag = tk*NT+tn ; per-frag layout [lane][e] with
//   K = tk*32 + (e/8)*16 + (lane/16)*8 + e%8 , N = tn*16 + (lane%16)   (ISA 05_wmma B layout)
__device__ __forceinline__ void stage_weight_frags(bf16_t* dst, const float* __restrict__ W,
                                                   int K, int N) {
    const int NT = N >> 4;
    const int total = K * N;
    for (int idx = threadIdx.x; idx < total; idx += THREADS) {
        int e    = idx & 15;
        int lane = (idx >> 4) & 31;
        int tile = idx >> 9;
        int tk   = tile / NT;
        int tn   = tile - tk * NT;
        int k    = tk * 32 + ((e >> 3) << 4) + ((lane >> 4) << 3) + (e & 7);
        int n    = tn * 16 + (lane & 15);
        dst[idx] = (bf16_t)W[n * K + k];
    }
}

__global__ __launch_bounds__(THREADS)
void gru_critic_kernel(const float* __restrict__ obs,
                       const float* __restrict__ h1, const float* __restrict__ h2,
                       const unsigned char* __restrict__ mask,
                       const float* __restrict__ Wf1_1, const float* __restrict__ bf1_1,
                       const float* __restrict__ wih_1, const float* __restrict__ whh_1,
                       const float* __restrict__ bih_1, const float* __restrict__ bhh_1,
                       const float* __restrict__ Wf2_1, const float* __restrict__ bf2_1,
                       const float* __restrict__ Wf1_2, const float* __restrict__ bf1_2,
                       const float* __restrict__ wih_2, const float* __restrict__ whh_2,
                       const float* __restrict__ bih_2, const float* __restrict__ bhh_2,
                       const float* __restrict__ Wf2_2, const float* __restrict__ bf2_2,
                       float* __restrict__ value1, float* __restrict__ value2,
                       float* __restrict__ nh1, float* __restrict__ nh2) {
    __shared__ __attribute__((aligned(32))) bf16_t wfrag[WFRAG_ELEMS];              // 96 KB
    __shared__ __attribute__((aligned(32))) bf16_t xbf[WAVES][2][16 * XH_STRIDE];   // 36 KB
    __shared__ __attribute__((aligned(32))) bf16_t hbf[WAVES][16 * XH_STRIDE];      // 18 KB

    const int tid  = threadIdx.x;
    const int wave = tid >> 5;
    const int lane = tid & 31;
    const int g    = lane >> 4;   // lane group (A-frag K half / C-frag M half)
    const int ln   = lane & 15;   // N index within tile / A-frag row
    const int row0 = blockIdx.x * ROWS_PER_BLOCK + wave * 16;

    // ---------------- Phase 1: stage fc1 weights (both streams) ----------------
    stage_weight_frags(wfrag,         Wf1_1, D_IN, H_DIM);
    stage_weight_frags(wfrag + 16384, Wf1_2, D_IN, H_DIM);
    __syncthreads();

    // ---------------- fc1: x_s = tanh(obs @ Wfc1_s^T + b) ----------------------
    v8f acc[2][4];
#pragma unroll
    for (int s = 0; s < 2; ++s)
#pragma unroll
        for (int tn = 0; tn < 4; ++tn)
            acc[s][tn] = (v8f){0.f, 0.f, 0.f, 0.f, 0.f, 0.f, 0.f, 0.f};

    const float* orow = obs + (size_t)(row0 + ln) * D_IN + g * 8;
#pragma unroll
    for (int tk = 0; tk < 8; ++tk) {
        const int k0 = tk * 32;
        // 16-bit A-frag (16x32): lane-group g reads K = k0+8g+e and k0+16+8g+e
        v4f f0 = *(const v4f*)(orow + k0);
        v4f f1 = *(const v4f*)(orow + k0 + 4);
        v4f f2 = *(const v4f*)(orow + k0 + 16);
        v4f f3 = *(const v4f*)(orow + k0 + 20);
        v16bf a;
#pragma unroll
        for (int i = 0; i < 4; ++i) {
            a[i]      = (bf16_t)f0[i];
            a[4 + i]  = (bf16_t)f1[i];
            a[8 + i]  = (bf16_t)f2[i];
            a[12 + i] = (bf16_t)f3[i];
        }
#pragma unroll
        for (int s = 0; s < 2; ++s) {
#pragma unroll
            for (int tn = 0; tn < 4; ++tn) {
                const v16bf b =
                    *(const v16bf*)&wfrag[s * 16384 + (((tk * 4 + tn) * 32 + lane) << 4)];
                acc[s][tn] = __builtin_amdgcn_wmma_f32_16x16x32_bf16(
                    false, a, false, b, (short)0, acc[s][tn], false, false);
            }
        }
    }

    {
        const float* bf1p[2] = {bf1_1, bf1_2};
#pragma unroll
        for (int s = 0; s < 2; ++s) {
#pragma unroll
            for (int tn = 0; tn < 4; ++tn) {
                const float bb = bf1p[s][tn * 16 + ln];
#pragma unroll
                for (int r = 0; r < 8; ++r) {
                    const float xv = tanh_f(acc[s][tn][r] + bb);
                    xbf[wave][s][(r + 8 * g) * XH_STRIDE + tn * 16 + ln] = (bf16_t)xv;
                }
            }
        }
    }
    __syncthreads();

    // ---------------- Phase 2: stage GRU weights (both streams) ----------------
    stage_weight_frags(wfrag,         wih_1, H_DIM, 192);
    stage_weight_frags(wfrag + 12288, wih_2, H_DIM, 192);
    stage_weight_frags(wfrag + 24576, whh_1, H_DIM, 192);
    stage_weight_frags(wfrag + 36864, whh_2, H_DIM, 192);
    __syncthreads();

    const float* hsrcA[2] = {h1, h2};
    const float* bihp[2]  = {bih_1, bih_2};
    const float* bhhp[2]  = {bhh_1, bhh_2};
    const float* wf2p[2]  = {Wf2_1, Wf2_2};
    const float* bf2p[2]  = {bf2_1, bf2_2};
    float*       nhout[2] = {nh1, nh2};
    float*       vout[2]  = {value1, value2};

    unsigned char mk[8];
#pragma unroll
    for (int r = 0; r < 8; ++r) mk[r] = mask[row0 + r + 8 * g];

#pragma unroll
    for (int s = 0; s < 2; ++s) {
        // Stage this stream's h tile (16 rows, contiguous 4 KB in global) as bf16.
        const float* hp = hsrcA[s] + (size_t)row0 * H_DIM;
        for (int i = lane; i < 16 * H_DIM; i += 32) {
            const int rr = i >> 6, cc = i & 63;
            hbf[wave][rr * XH_STRIDE + cc] = (bf16_t)hp[i];
        }
        __syncthreads();

        const bf16_t* ihbase = wfrag + s * 12288;
        const bf16_t* hhbase = wfrag + 24576 + s * 12288;

        float p[8];
#pragma unroll
        for (int r = 0; r < 8; ++r) p[r] = 0.f;

#pragma unroll
        for (int t = 0; t < 4; ++t) {
            v8f ir, iz, in_, hr, hz, hn;
            ir = iz = in_ = hr = hz = hn = (v8f){0.f, 0.f, 0.f, 0.f, 0.f, 0.f, 0.f, 0.f};
#pragma unroll
            for (int tk = 0; tk < 2; ++tk) {
                const int k0 = tk * 32;
                const bf16_t* xb = &xbf[wave][s][ln * XH_STRIDE + k0 + 8 * g];
                v8bf xlo = *(const v8bf*)xb;
                v8bf xhi = *(const v8bf*)(xb + 16);
                v16bf ax = __builtin_shufflevector(xlo, xhi, 0, 1, 2, 3, 4, 5, 6, 7,
                                                   8, 9, 10, 11, 12, 13, 14, 15);
                const bf16_t* hb = &hbf[wave][ln * XH_STRIDE + k0 + 8 * g];
                v8bf hlo = *(const v8bf*)hb;
                v8bf hhi = *(const v8bf*)(hb + 16);
                v16bf ah = __builtin_shufflevector(hlo, hhi, 0, 1, 2, 3, 4, 5, 6, 7,
                                                   8, 9, 10, 11, 12, 13, 14, 15);
                v16bf b;
                b = *(const v16bf*)&ihbase[((tk * 12 + t) * 32 + lane) * 16];
                ir = __builtin_amdgcn_wmma_f32_16x16x32_bf16(false, ax, false, b, (short)0, ir, false, false);
                b = *(const v16bf*)&ihbase[((tk * 12 + t + 4) * 32 + lane) * 16];
                iz = __builtin_amdgcn_wmma_f32_16x16x32_bf16(false, ax, false, b, (short)0, iz, false, false);
                b = *(const v16bf*)&ihbase[((tk * 12 + t + 8) * 32 + lane) * 16];
                in_ = __builtin_amdgcn_wmma_f32_16x16x32_bf16(false, ax, false, b, (short)0, in_, false, false);
                b = *(const v16bf*)&hhbase[((tk * 12 + t) * 32 + lane) * 16];
                hr = __builtin_amdgcn_wmma_f32_16x16x32_bf16(false, ah, false, b, (short)0, hr, false, false);
                b = *(const v16bf*)&hhbase[((tk * 12 + t + 4) * 32 + lane) * 16];
                hz = __builtin_amdgcn_wmma_f32_16x16x32_bf16(false, ah, false, b, (short)0, hz, false, false);
                b = *(const v16bf*)&hhbase[((tk * 12 + t + 8) * 32 + lane) * 16];
                hn = __builtin_amdgcn_wmma_f32_16x16x32_bf16(false, ah, false, b, (short)0, hn, false, false);
            }
            // Fused GRU elementwise + mask blend + fc2 partial dot.
            const int   ncol = t * 16 + ln;
            const float bir = bihp[s][ncol], biz = bihp[s][64 + ncol], bin = bihp[s][128 + ncol];
            const float bhr = bhhp[s][ncol], bhz = bhhp[s][64 + ncol], bhn = bhhp[s][128 + ncol];
            const float w2  = wf2p[s][ncol];
#pragma unroll
            for (int r = 0; r < 8; ++r) {
                const int   mrow = r + 8 * g;
                const float hold = hp[(size_t)mrow * H_DIM + ncol];
                const float rg = sigmoid_f(ir[r] + hr[r] + bir + bhr);
                const float zg = sigmoid_f(iz[r] + hz[r] + biz + bhz);
                const float ng = tanh_f(in_[r] + bin + rg * (hn[r] + bhn));
                const float nh = mk[r] ? ((1.f - zg) * ng + zg * hold) : hold;
                nhout[s][(size_t)(row0 + mrow) * H_DIM + ncol] = nh;
                p[r] += nh * w2;
            }
        }
        // fc2: reduce over the 16 lanes of each half-wave (rows r+8g).
        const float b2 = bf2p[s][0];
#pragma unroll
        for (int r = 0; r < 8; ++r) {
            float v = p[r];
            v += __shfl_xor(v, 8, 32);
            v += __shfl_xor(v, 4, 32);
            v += __shfl_xor(v, 2, 32);
            v += __shfl_xor(v, 1, 32);
            if (ln == 0) vout[s][row0 + r + 8 * g] = v + b2;
        }
        __syncthreads();
    }
}

extern "C" void kernel_launch(void* const* d_in, const int* in_sizes, int n_in,
                              void* d_out, int out_size, void* d_ws, size_t ws_size,
                              hipStream_t stream) {
    const float* obs = (const float*)d_in[0];
    const float* h1  = (const float*)d_in[1];
    const float* h2  = (const float*)d_in[2];
    const unsigned char* mask = (const unsigned char*)d_in[3];  // jax bool -> 1 byte/elem
    const float* Wf1_1 = (const float*)d_in[4];
    const float* bf1_1 = (const float*)d_in[5];
    const float* wih_1 = (const float*)d_in[6];
    const float* whh_1 = (const float*)d_in[7];
    const float* bih_1 = (const float*)d_in[8];
    const float* bhh_1 = (const float*)d_in[9];
    const float* Wf2_1 = (const float*)d_in[10];
    const float* bf2_1 = (const float*)d_in[11];
    const float* Wf1_2 = (const float*)d_in[12];
    const float* bf1_2 = (const float*)d_in[13];
    const float* wih_2 = (const float*)d_in[14];
    const float* whh_2 = (const float*)d_in[15];
    const float* bih_2 = (const float*)d_in[16];
    const float* bhh_2 = (const float*)d_in[17];
    const float* Wf2_2 = (const float*)d_in[18];
    const float* bf2_2 = (const float*)d_in[19];

    float* out    = (float*)d_out;
    float* value1 = out;
    float* value2 = out + B_TOTAL;
    float* nh1    = out + 2 * (size_t)B_TOTAL;
    float* nh2    = nh1 + (size_t)B_TOTAL * H_DIM;

    dim3 grid(B_TOTAL / ROWS_PER_BLOCK);
    dim3 block(THREADS);
    hipLaunchKernelGGL(gru_critic_kernel, grid, block, 0, stream,
                       obs, h1, h2, mask,
                       Wf1_1, bf1_1, wih_1, whh_1, bih_1, bhh_1, Wf2_1, bf2_1,
                       Wf1_2, bf1_2, wih_2, whh_2, bih_2, bhh_2, Wf2_2, bf2_2,
                       value1, value2, nh1, nh2);
}
